// VectorMessagePassingLayer_77335181132461
// MI455X (gfx1250) — matrix operand order repaired and checked by reading, not compiled, not run on hardware
//
#include <hip/hip_runtime.h>

typedef __attribute__((ext_vector_type(16))) __bf16 v16bf;
typedef __attribute__((ext_vector_type(8)))  __bf16 v8bf;
typedef __attribute__((ext_vector_type(8)))  float  v8f;

#define HIDDEN 128
#define NB 2          // batch
#define LNEPS 1e-5f
#define EMBC 0.3289407f   // ln(10)/7

__device__ __forceinline__ __bf16 f2bf(float f) {
    unsigned u = __float_as_uint(f);
    unsigned r = u + 0x7FFFu + ((u >> 16) & 1u);
    unsigned short s = (unsigned short)(r >> 16);
    __bf16 b;
    __builtin_memcpy(&b, &s, 2);
    return b;
}

__device__ __forceinline__ void atomicMaxF(float* addr, float val) {
    int* ai = (int*)addr;
    int old = __float_as_int(*addr);
    while (__int_as_float(old) < val) {
        int assumed = old;
        old = atomicCAS(ai, assumed, __float_as_int(val));
        if (old == assumed) break;
    }
}

// ---------------------------------------------------------------- init
__global__ void init_kernel(float* nmax, float* nsum, float* agg,
                            int nMaxSum, int nAgg) {
    int i = blockIdx.x * blockDim.x + threadIdx.x;
    if (i < nMaxSum) { nmax[i] = -3.402823466e38f; nsum[i] = 0.f; }
    for (int j = i; j < nAgg; j += gridDim.x * blockDim.x) agg[j] = 0.f;
}

// ---------------------------------------------------------------- logits + segment max
__global__ void logits_kernel(const float* __restrict__ x, const float* __restrict__ pos,
                              const float* __restrict__ Wa, const float* __restrict__ ba,
                              const long long* __restrict__ ei,
                              float* __restrict__ logits, float* __restrict__ nmax,
                              int V, int E) {
    __shared__ float sWa[288 * 4 + 4];
    for (int i = threadIdx.x; i < 288 * 4; i += blockDim.x) sWa[i] = Wa[i];
    if (threadIdx.x < 4) sWa[288 * 4 + threadIdx.x] = ba[threadIdx.x];
    __syncthreads();

    int idx = blockIdx.x * blockDim.x + threadIdx.x;
    if (idx >= NB * E) return;
    int b = idx / E;
    int e = idx - b * E;
    int src = (int)ei[e];
    int dst = (int)ei[E + e];
    const float* xs = x + ((size_t)b * V + src) * HIDDEN;
    const float* xd = x + ((size_t)b * V + dst) * HIDDEN;
    float rx = pos[((size_t)b * V + src) * 2 + 0] - pos[((size_t)b * V + dst) * 2 + 0];
    float ry = pos[((size_t)b * V + src) * 2 + 1] - pos[((size_t)b * V + dst) * 2 + 1];
    float a0 = sWa[1152], a1 = sWa[1153], a2 = sWa[1154], a3 = sWa[1155];
    for (int c = 0; c < 288; ++c) {
        float mi;
        if (c < 128)       mi = xs[c];
        else if (c < 256)  mi = xd[c - 128];
        else {
            int cc = c - 256; int j = cc & 7; int g = cc >> 3;
            float fr = __expf(-(float)j * EMBC);
            float ang = ((g < 2) ? rx : ry) * fr;
            mi = ((g & 1) == 0) ? __sinf(ang) : __cosf(ang);
        }
        a0 += mi * sWa[c * 4 + 0];
        a1 += mi * sWa[c * 4 + 1];
        a2 += mi * sWa[c * 4 + 2];
        a3 += mi * sWa[c * 4 + 3];
    }
    size_t lo = (size_t)idx * 4;
    logits[lo + 0] = a0; logits[lo + 1] = a1; logits[lo + 2] = a2; logits[lo + 3] = a3;
    float* nm = nmax + ((size_t)b * V + dst) * 4;
    atomicMaxF(nm + 0, a0); atomicMaxF(nm + 1, a1);
    atomicMaxF(nm + 2, a2); atomicMaxF(nm + 3, a3);
}

// ---------------------------------------------------------------- exp + segment sum
__global__ void expsum_kernel(const float* __restrict__ logits, const float* __restrict__ nmax,
                              float* __restrict__ nsum, const long long* __restrict__ ei,
                              float* __restrict__ exbuf, int V, int E) {
    int idx = blockIdx.x * blockDim.x + threadIdx.x;
    if (idx >= NB * E * 4) return;
    int h = idx & 3;
    int t = idx >> 2;
    int b = t / E;
    int e = t - b * E;
    int dst = (int)ei[E + e];
    float ex = __expf(logits[idx] - nmax[((size_t)b * V + dst) * 4 + h]);
    exbuf[idx] = ex;
    atomicAdd(&nsum[((size_t)b * V + dst) * 4 + h], ex);
}

// ---------------------------------------------------------------- edge message MLP (WMMA) + scatter
__global__ void __launch_bounds__(256, 1)
edge_msg_kernel(const float* __restrict__ x, const float* __restrict__ pos,
                const float* __restrict__ W1, const float* __restrict__ b1,
                const float* __restrict__ g1, const float* __restrict__ be1,
                const float* __restrict__ W2, const float* __restrict__ b2,
                const long long* __restrict__ ei, const float* __restrict__ exbuf,
                const float* __restrict__ nsum, float* __restrict__ agg,
                int V, int E) {
    __shared__ __align__(16) __bf16 sA[16 * 288];
    __shared__ __align__(16) __bf16 sH[16 * 256];
    __shared__ float sC[16 * 128];
    __shared__ float sPS[16 * 8], sPQ[16 * 8];
    __shared__ float sMu[16], sRs[16];
    __shared__ float sB1[256], sG1[256], sBe1[256], sB2[128];
    __shared__ int   sSrc[16], sDst[16];

    const int tid  = threadIdx.x;
    const int lane = tid & 31;
    const int wave = tid >> 5;
    const int hl   = lane >> 4;     // lane half
    const int l15  = lane & 15;

    sB1[tid & 255] = b1[tid & 255]; sG1[tid & 255] = g1[tid & 255]; sBe1[tid & 255] = be1[tid & 255];
    if (tid < 128) sB2[tid] = b2[tid];

    // -------- register-resident weight fragments (loaded once, amortized) --------
    v16bf w1f[9][2];
#pragma unroll
    for (int kt = 0; kt < 9; ++kt)
#pragma unroll
        for (int t = 0; t < 2; ++t) {
            int n = (2 * wave + t) * 16 + l15;
            int kbase = kt * 32 + hl * 16;
            v16bf f;
#pragma unroll
            for (int i = 0; i < 16; ++i) f[i] = f2bf(W1[(size_t)(kbase + i) * 256 + n]);
            w1f[kt][t] = f;
        }
    v16bf w2f[8];
#pragma unroll
    for (int nt = 0; nt < 8; ++nt) {
        int n = nt * 16 + l15;
        int kbase = wave * 32 + hl * 16;
        v16bf f;
#pragma unroll
        for (int i = 0; i < 16; ++i) f[i] = f2bf(W2[(size_t)(kbase + i) * 128 + n]);
        w2f[nt] = f;
    }
    __syncthreads();

    const int nJobs = (E / 16) * NB;
    for (int job = blockIdx.x; job < nJobs; job += gridDim.x) {
        const int b  = job & 1;
        const int e0 = (job >> 1) * 16;

        if (tid < 16) { sSrc[tid] = (int)ei[e0 + tid]; sDst[tid] = (int)ei[E + e0 + tid]; }
#pragma unroll
        for (int i = 0; i < 8; ++i) sC[tid + 256 * i] = 0.f;
        __syncthreads();

        // build msg_in tile (16 x 288) in bf16
        for (int i = tid; i < 16 * 288; i += 256) {
            int m2 = i / 288, c = i - m2 * 288;
            int srcN = sSrc[m2], dstN = sDst[m2];
            float v;
            if (c < 128)      v = x[((size_t)b * V + srcN) * HIDDEN + c];
            else if (c < 256) v = x[((size_t)b * V + dstN) * HIDDEN + (c - 128)];
            else {
                int cc = c - 256; int j = cc & 7; int g = cc >> 3;
                float fr = __expf(-(float)j * EMBC);
                float rel = (g < 2)
                    ? (pos[((size_t)b * V + srcN) * 2 + 0] - pos[((size_t)b * V + dstN) * 2 + 0])
                    : (pos[((size_t)b * V + srcN) * 2 + 1] - pos[((size_t)b * V + dstN) * 2 + 1]);
                float ang = rel * fr;
                v = ((g & 1) == 0) ? __sinf(ang) : __cosf(ang);
            }
            sA[m2 * 288 + c] = f2bf(v);
        }
        __syncthreads();

        // ---- layer 1: (16x288) @ (288x256), wave owns 32 output cols ----
        v8f acc0 = {0,0,0,0,0,0,0,0}, acc1 = acc0;
        const int m = l15;
#pragma unroll
        for (int kt = 0; kt < 9; ++kt) {
            int koff = kt * 32 + hl * 16;
            v8bf lo = *(const v8bf*)(sA + m * 288 + koff);
            v8bf hi = *(const v8bf*)(sA + m * 288 + koff + 8);
            v16bf a;
#pragma unroll
            for (int i = 0; i < 8; ++i) { a[i] = lo[i]; a[i + 8] = hi[i]; }
            acc0 = __builtin_amdgcn_wmma_f32_16x16x32_bf16(false, a, false, w1f[kt][0], (short)0, acc0, false, false);
            acc1 = __builtin_amdgcn_wmma_f32_16x16x32_bf16(false, a, false, w1f[kt][1], (short)0, acc1, false, false);
        }
        const int n0 = (2 * wave) * 16 + l15;
        const int n1 = n0 + 16;
#pragma unroll
        for (int r = 0; r < 8; ++r) { acc0[r] += sB1[n0]; acc1[r] += sB1[n1]; }

        // ---- cross-wave LayerNorm over 256 ----
#pragma unroll
        for (int r = 0; r < 8; ++r) {
            float s = acc0[r] + acc1[r];
            float q = acc0[r] * acc0[r] + acc1[r] * acc1[r];
#pragma unroll
            for (int off = 1; off < 16; off <<= 1) { s += __shfl_xor(s, off, 32); q += __shfl_xor(q, off, 32); }
            if (l15 == 0) { int row = r + 8 * hl; sPS[row * 8 + wave] = s; sPQ[row * 8 + wave] = q; }
        }
        __syncthreads();
        if (tid < 16) {
            float s = 0.f, q = 0.f;
#pragma unroll
            for (int w = 0; w < 8; ++w) { s += sPS[tid * 8 + w]; q += sPQ[tid * 8 + w]; }
            float mu = s * (1.f / 256.f);
            sMu[tid] = mu;
            sRs[tid] = rsqrtf(q * (1.f / 256.f) - mu * mu + LNEPS);
        }
        __syncthreads();
#pragma unroll
        for (int r = 0; r < 8; ++r) {
            int row = r + 8 * hl;
            float mu = sMu[row], rs = sRs[row];
            float h0 = (acc0[r] - mu) * rs * sG1[n0] + sBe1[n0];
            float h1 = (acc1[r] - mu) * rs * sG1[n1] + sBe1[n1];
            h0 = h0 / (1.f + __expf(-h0));
            h1 = h1 / (1.f + __expf(-h1));
            sH[row * 256 + n0] = f2bf(h0);
            sH[row * 256 + n1] = f2bf(h1);
        }
        __syncthreads();

        // ---- layer 2: K split across waves, LDS f32 reduce ----
        {
            int koff = wave * 32 + hl * 16;
            v8bf lo = *(const v8bf*)(sH + m * 256 + koff);
            v8bf hi = *(const v8bf*)(sH + m * 256 + koff + 8);
            v16bf a;
#pragma unroll
            for (int i = 0; i < 8; ++i) { a[i] = lo[i]; a[i + 8] = hi[i]; }
#pragma unroll
            for (int nt = 0; nt < 8; ++nt) {
                v8f c = {0,0,0,0,0,0,0,0};
                c = __builtin_amdgcn_wmma_f32_16x16x32_bf16(false, a, false, w2f[nt], (short)0, c, false, false);
                int n = nt * 16 + l15;
#pragma unroll
                for (int r = 0; r < 8; ++r) atomicAdd(&sC[(r + 8 * hl) * 128 + n], c[r]);
            }
        }
        __syncthreads();

        // ---- attention weight + global scatter-add ----
        {
            int row = tid >> 4;
            int cb  = (tid & 15) * 8;
            int e   = e0 + row;
            int dst = sDst[row];
            int head = (tid & 15) >> 2;   // cb>>5
            float ex = exbuf[((size_t)b * E + e) * 4 + head];
            float sm = nsum[((size_t)b * V + dst) * 4 + head];
            float aw = ex / (sm + 1e-8f);
            float* ag = agg + ((size_t)b * V + dst) * HIDDEN;
#pragma unroll
            for (int j = 0; j < 8; ++j) {
                int c = cb + j;
                atomicAdd(&ag[c], (sC[row * 128 + c] + sB2[c]) * aw);
            }
        }
        __syncthreads();
    }
}

// ---------------------------------------------------------------- node update MLP (WMMA)
__global__ void __launch_bounds__(256, 1)
node_update_kernel(const float* __restrict__ x, const float* __restrict__ agg,
                   const float* __restrict__ Wu1, const float* __restrict__ bu1,
                   const float* __restrict__ gu, const float* __restrict__ beu,
                   const float* __restrict__ Wu2, const float* __restrict__ bu2,
                   const float* __restrict__ gf, const float* __restrict__ bfv,
                   float* __restrict__ out, int V) {
    __shared__ __align__(16) __bf16 sA[16 * 256];
    __shared__ __align__(16) __bf16 sH[16 * 256];
    __shared__ float sC[16 * 128];
    __shared__ float sPS[16 * 8], sPQ[16 * 8];
    __shared__ float sMu[16], sRs[16];
    __shared__ float sFS[16], sFQ[16];
    __shared__ float sBu1[256], sGu[256], sBeu[256];
    __shared__ float sBu2[128], sGf[128], sBf[128];

    const int tid  = threadIdx.x;
    const int lane = tid & 31;
    const int wave = tid >> 5;
    const int hl   = lane >> 4;
    const int l15  = lane & 15;

    sBu1[tid & 255] = bu1[tid & 255]; sGu[tid & 255] = gu[tid & 255]; sBeu[tid & 255] = beu[tid & 255];
    if (tid < 128) { sBu2[tid] = bu2[tid]; sGf[tid] = gf[tid]; sBf[tid] = bfv[tid]; }

    v16bf w1f[8][2];
#pragma unroll
    for (int kt = 0; kt < 8; ++kt)
#pragma unroll
        for (int t = 0; t < 2; ++t) {
            int n = (2 * wave + t) * 16 + l15;
            int kbase = kt * 32 + hl * 16;
            v16bf f;
#pragma unroll
            for (int i = 0; i < 16; ++i) f[i] = f2bf(Wu1[(size_t)(kbase + i) * 256 + n]);
            w1f[kt][t] = f;
        }
    v16bf w2f[8];
#pragma unroll
    for (int nt = 0; nt < 8; ++nt) {
        int n = nt * 16 + l15;
        int kbase = wave * 32 + hl * 16;
        v16bf f;
#pragma unroll
        for (int i = 0; i < 16; ++i) f[i] = f2bf(Wu2[(size_t)(kbase + i) * 128 + n]);
        w2f[nt] = f;
    }
    __syncthreads();

    const int nJobs = (V / 16) * NB;
    for (int job = blockIdx.x; job < nJobs; job += gridDim.x) {
        const int b  = job & 1;
        const int v0 = (job >> 1) * 16;

#pragma unroll
        for (int i = 0; i < 8; ++i) sC[tid + 256 * i] = 0.f;
        if (tid < 16) { sFS[tid] = 0.f; sFQ[tid] = 0.f; }
        for (int i = tid; i < 16 * 256; i += 256) {
            int m2 = i >> 8, c = i & 255;
            size_t base = ((size_t)b * V + v0 + m2) * HIDDEN;
            float v = (c < 128) ? x[base + c] : agg[base + (c - 128)];
            sA[i] = f2bf(v);
        }
        __syncthreads();

        v8f acc0 = {0,0,0,0,0,0,0,0}, acc1 = acc0;
        const int m = l15;
#pragma unroll
        for (int kt = 0; kt < 8; ++kt) {
            int koff = kt * 32 + hl * 16;
            v8bf lo = *(const v8bf*)(sA + m * 256 + koff);
            v8bf hi = *(const v8bf*)(sA + m * 256 + koff + 8);
            v16bf a;
#pragma unroll
            for (int i = 0; i < 8; ++i) { a[i] = lo[i]; a[i + 8] = hi[i]; }
            acc0 = __builtin_amdgcn_wmma_f32_16x16x32_bf16(false, a, false, w1f[kt][0], (short)0, acc0, false, false);
            acc1 = __builtin_amdgcn_wmma_f32_16x16x32_bf16(false, a, false, w1f[kt][1], (short)0, acc1, false, false);
        }
        const int n0 = (2 * wave) * 16 + l15;
        const int n1 = n0 + 16;
#pragma unroll
        for (int r = 0; r < 8; ++r) { acc0[r] += sBu1[n0]; acc1[r] += sBu1[n1]; }

#pragma unroll
        for (int r = 0; r < 8; ++r) {
            float s = acc0[r] + acc1[r];
            float q = acc0[r] * acc0[r] + acc1[r] * acc1[r];
#pragma unroll
            for (int off = 1; off < 16; off <<= 1) { s += __shfl_xor(s, off, 32); q += __shfl_xor(q, off, 32); }
            if (l15 == 0) { int row = r + 8 * hl; sPS[row * 8 + wave] = s; sPQ[row * 8 + wave] = q; }
        }
        __syncthreads();
        if (tid < 16) {
            float s = 0.f, q = 0.f;
#pragma unroll
            for (int w = 0; w < 8; ++w) { s += sPS[tid * 8 + w]; q += sPQ[tid * 8 + w]; }
            float mu = s * (1.f / 256.f);
            sMu[tid] = mu;
            sRs[tid] = rsqrtf(q * (1.f / 256.f) - mu * mu + LNEPS);
        }
        __syncthreads();
#pragma unroll
        for (int r = 0; r < 8; ++r) {
            int row = r + 8 * hl;
            float mu = sMu[row], rs = sRs[row];
            float h0 = (acc0[r] - mu) * rs * sGu[n0] + sBeu[n0];
            float h1 = (acc1[r] - mu) * rs * sGu[n1] + sBeu[n1];
            h0 = h0 / (1.f + __expf(-h0));
            h1 = h1 / (1.f + __expf(-h1));
            sH[row * 256 + n0] = f2bf(h0);
            sH[row * 256 + n1] = f2bf(h1);
        }
        __syncthreads();

        {
            int koff = wave * 32 + hl * 16;
            v8bf lo = *(const v8bf*)(sH + m * 256 + koff);
            v8bf hi = *(const v8bf*)(sH + m * 256 + koff + 8);
            v16bf a;
#pragma unroll
            for (int i = 0; i < 8; ++i) { a[i] = lo[i]; a[i + 8] = hi[i]; }
#pragma unroll
            for (int nt = 0; nt < 8; ++nt) {
                v8f c = {0,0,0,0,0,0,0,0};
                c = __builtin_amdgcn_wmma_f32_16x16x32_bf16(false, a, false, w2f[nt], (short)0, c, false, false);
                int n = nt * 16 + l15;
#pragma unroll
                for (int r = 0; r < 8; ++r) atomicAdd(&sC[(r + 8 * hl) * 128 + n], c[r]);
            }
        }
        __syncthreads();

        // residual + final LayerNorm(128) + store
        {
            int row = tid >> 4;
            int cb  = (tid & 15) * 8;
            size_t base = ((size_t)b * V + v0 + row) * HIDDEN;
            float v[8]; float s = 0.f, q = 0.f;
#pragma unroll
            for (int j = 0; j < 8; ++j) {
                float t2 = sC[row * 128 + cb + j] + sBu2[cb + j] + x[base + cb + j];
                v[j] = t2; s += t2; q += t2 * t2;
            }
            atomicAdd(&sFS[row], s);
            atomicAdd(&sFQ[row], q);
            __syncthreads();
            float mu = sFS[row] * (1.f / 128.f);
            float rs = rsqrtf(sFQ[row] * (1.f / 128.f) - mu * mu + LNEPS);
#pragma unroll
            for (int j = 0; j < 8; ++j)
                out[base + cb + j] = (v[j] - mu) * rs * sGf[cb + j] + sBf[cb + j];
        }
        __syncthreads();
    }
}

// ---------------------------------------------------------------- launch
extern "C" void kernel_launch(void* const* d_in, const int* in_sizes, int n_in,
                              void* d_out, int out_size, void* d_ws, size_t ws_size,
                              hipStream_t stream) {
    const float* x    = (const float*)d_in[0];
    const float* pos  = (const float*)d_in[1];
    const float* W1m  = (const float*)d_in[2];
    const float* b1m  = (const float*)d_in[3];
    const float* g1m  = (const float*)d_in[4];
    const float* be1m = (const float*)d_in[5];
    const float* W2m  = (const float*)d_in[6];
    const float* b2m  = (const float*)d_in[7];
    const float* Wa   = (const float*)d_in[8];
    const float* ba   = (const float*)d_in[9];
    const float* Wu1  = (const float*)d_in[10];
    const float* bu1  = (const float*)d_in[11];
    const float* gu   = (const float*)d_in[12];
    const float* beu  = (const float*)d_in[13];
    const float* Wu2  = (const float*)d_in[14];
    const float* bu2  = (const float*)d_in[15];
    const float* gf   = (const float*)d_in[16];
    const float* bf   = (const float*)d_in[17];
    const long long* ei = (const long long*)d_in[18];
    float* out = (float*)d_out;

    const int E = in_sizes[18] / 2;
    const int V = in_sizes[0] / (NB * HIDDEN);

    float* logits = (float*)d_ws;                    // B*E*4
    float* exbuf  = logits + (size_t)NB * E * 4;     // B*E*4
    float* nmax   = exbuf  + (size_t)NB * E * 4;     // B*V*4
    float* nsum   = nmax   + (size_t)NB * V * 4;     // B*V*4
    float* agg    = nsum   + (size_t)NB * V * 4;     // B*V*128

    const int nMaxSum = NB * V * 4;
    const int nAgg    = NB * V * HIDDEN;
    init_kernel<<<(nAgg + 255) / 256, 256, 0, stream>>>(nmax, nsum, agg, nMaxSum, nAgg);

    const int nE = NB * E;
    logits_kernel<<<(nE + 255) / 256, 256, 0, stream>>>(x, pos, Wa, ba, ei, logits, nmax, V, E);

    const int nEh = NB * E * 4;
    expsum_kernel<<<(nEh + 255) / 256, 256, 0, stream>>>(logits, nmax, nsum, ei, exbuf, V, E);

    edge_msg_kernel<<<2048, 256, 0, stream>>>(x, pos, W1m, b1m, g1m, be1m, W2m, b2m,
                                              ei, exbuf, nsum, agg, V, E);

    node_update_kernel<<<(V / 16) * NB, 256, 0, stream>>>(x, agg, Wu1, bu1, gu, beu,
                                                          Wu2, bu2, gf, bf, out, V);
}
